// GPTNeoGQASelfAttention_23570780520712
// MI455X (gfx1250) — matrix operand order repaired and checked by reading, not compiled, not run on hardware
//
#include <hip/hip_runtime.h>
#include <hip/hip_bf16.h>

typedef __bf16 bf16;
typedef __attribute__((ext_vector_type(16))) __bf16 bf16x16;
typedef __attribute__((ext_vector_type(8)))  __bf16 bf16x8;
typedef __attribute__((ext_vector_type(4)))  __bf16 bf16x4;
typedef __attribute__((ext_vector_type(8)))  float  f32x8;

#define S_LEN 4096
#define DMODEL 1024
#define NHEADS 16
#define HDIM 64
#define WIN 256
#define MASK_NEG (-3.0e38f)

#define WMMA_BF16(a, b, c) \
  __builtin_amdgcn_wmma_f32_16x16x32_bf16(false, (a), false, (b), (short)0, (c), false, false)

// ---------------------------------------------------------------------------
// Fragment loaders (layouts per CDNA5 ISA §7.12.2)
// A 16x32 bf16: lane m = lane&15 ; K = {0..7,16..23} (lanes 0-15) / {8..15,24..31}
// ---------------------------------------------------------------------------
__device__ __forceinline__ bf16x16 load_a_frag(const bf16* __restrict__ t, int ld) {
  const int lane = threadIdx.x & 31;
  const bf16* p = t + (lane & 15) * ld + ((lane >> 4) << 3);
  bf16x8 lo = *(const bf16x8*)(p);
  bf16x8 hi = *(const bf16x8*)(p + 16);
  return __builtin_shufflevector(lo, hi, 0,1,2,3,4,5,6,7,8,9,10,11,12,13,14,15);
}

// B 32x16 bf16 from an N-major (i.e. B-transposed row-major) source:
// lane n = lane&15 ; K = {0..7,16..23} (lanes 0-15) / {8..15,24..31} (lanes 16-31).
__device__ __forceinline__ bf16x16 load_b_frag(const bf16* __restrict__ t, int ld) {
  const int lane = threadIdx.x & 31;
  const bf16* p = t + (lane & 15) * ld + ((lane >> 4) << 3);
  bf16x8 lo = *(const bf16x8*)(p);
  bf16x8 hi = *(const bf16x8*)(p + 16);
  return __builtin_shufflevector(lo, hi, 0,1,2,3,4,5,6,7,8,9,10,11,12,13,14,15);
}

// v_permlane16_b32 butterfly (VALU, per-16-lane row; replaces ds_bpermute)
__device__ __forceinline__ float permxor(float v, unsigned lo, unsigned hi) {
  union { float f; unsigned u; } a, r;
  a.f = v;
  r.u = __builtin_amdgcn_permlane16(a.u, a.u, lo, hi, false, false);
  return r.f;
}
// nibble-select words: sel[i] = i ^ step (lanes 0-7 in lo, 8-15 in hi)
#define X1LO 0x67452301u
#define X1HI 0xEFCDAB89u
#define X2LO 0x54761032u
#define X2HI 0xDCFE98BAu
#define X4LO 0x32107654u
#define X4HI 0xBA98FEDCu
#define X8LO 0xFEDCBA98u
#define X8HI 0x76543210u

__device__ __forceinline__ float rowmax16(float t) {
  t = fmaxf(t, permxor(t, X1LO, X1HI));
  t = fmaxf(t, permxor(t, X2LO, X2HI));
  t = fmaxf(t, permxor(t, X4LO, X4HI));
  t = fmaxf(t, permxor(t, X8LO, X8HI));
  return t;
}

// ---------------------------------------------------------------------------
// fp32 -> bf16 convert (vectorized x4)
// ---------------------------------------------------------------------------
__global__ void cvt_f32_bf16(const float* __restrict__ x, bf16* __restrict__ y, int n4) {
  int i = blockIdx.x * blockDim.x + threadIdx.x;
  if (i < n4) {
    float4 v = ((const float4*)x)[i];
    bf16x4 o;
    o[0] = (bf16)v.x; o[1] = (bf16)v.y; o[2] = (bf16)v.z; o[3] = (bf16)v.w;
    ((bf16x4*)y)[i] = o;
  }
}

// ---------------------------------------------------------------------------
// W[k][n] f32 (1024x1024) -> Wt[n][k] bf16, LDS-tiled transpose
// ---------------------------------------------------------------------------
__global__ void transpose_cvt(const float* __restrict__ W, bf16* __restrict__ Wt) {
  __shared__ float t[32][33];
  int x = blockIdx.x * 32 + threadIdx.x;
  int y0 = blockIdx.y * 32;
  #pragma unroll
  for (int r = threadIdx.y; r < 32; r += 8)
    t[r][threadIdx.x] = W[(y0 + r) * DMODEL + x];
  __syncthreads();
  int xo  = blockIdx.y * 32 + threadIdx.x;
  int yo0 = blockIdx.x * 32;
  #pragma unroll
  for (int r = threadIdx.y; r < 32; r += 8)
    Wt[(yo0 + r) * DMODEL + xo] = (bf16)t[threadIdx.x][r];
}

// ---------------------------------------------------------------------------
// C[4096,1024] = A[4096,1024] * Bt^T   (A bf16 row-major, Bt = B^T bf16 [N][K])
// MODE 0: store bf16 row-major [M,N]
// MODE 1: store bf16 transposed [N,M]  (contiguous-in-M -> packed b128 stores)
// MODE 2: store f32 [M,N] + bias[N]    (final out_proj)
// Block: 256 thr (8 waves); block tile 128x128; wave tile 32x64 (2x4 C frags)
// ---------------------------------------------------------------------------
template <int MODE>
__global__ __launch_bounds__(256) void gemm_bf16(const bf16* __restrict__ A,
                                                 const bf16* __restrict__ Bt,
                                                 void* __restrict__ Out,
                                                 const float* __restrict__ bias) {
  const int lane = threadIdx.x & 31;
  const int wv   = threadIdx.x >> 5;
  const int m0 = blockIdx.y * 128 + (wv & 3) * 32;
  const int n0 = blockIdx.x * 128 + (wv >> 2) * 64;

  f32x8 c[2][4];
  #pragma unroll
  for (int i = 0; i < 2; ++i)
    #pragma unroll
    for (int j = 0; j < 4; ++j)
      #pragma unroll
      for (int e = 0; e < 8; ++e) c[i][j][e] = 0.f;

  for (int k0 = 0; k0 < DMODEL; k0 += 32) {
    __builtin_prefetch(A + m0 * DMODEL + k0 + 128, 0, 1);
    bf16x16 a0 = load_a_frag(A + m0 * DMODEL + k0, DMODEL);
    bf16x16 a1 = load_a_frag(A + (m0 + 16) * DMODEL + k0, DMODEL);
    #pragma unroll
    for (int j = 0; j < 4; ++j) {
      bf16x16 b = load_b_frag(Bt + (n0 + j * 16) * DMODEL + k0, DMODEL);
      c[0][j] = WMMA_BF16(a0, b, c[0][j]);
      c[1][j] = WMMA_BF16(a1, b, c[1][j]);
    }
  }

  const int half = lane >> 4, col = lane & 15;
  #pragma unroll
  for (int i = 0; i < 2; ++i)
    #pragma unroll
    for (int j = 0; j < 4; ++j) {
      if (MODE == 1) {
        // transposed out: a lane's 8 slots are 8 consecutive M at fixed N
        int N = n0 + j * 16 + col;
        int Mb = m0 + i * 16 + half * 8;
        bf16x8 v;
        #pragma unroll
        for (int r = 0; r < 8; ++r) v[r] = (bf16)c[i][j][r];
        *(bf16x8*)((bf16*)Out + (size_t)N * S_LEN + Mb) = v;
      } else {
        #pragma unroll
        for (int r = 0; r < 8; ++r) {
          int M = m0 + i * 16 + r + half * 8;
          int N = n0 + j * 16 + col;
          float v = c[i][j][r];
          if (MODE == 0) ((bf16*)Out)[M * DMODEL + N] = (bf16)v;
          else           ((float*)Out)[M * DMODEL + N] = v + bias[N];
        }
      }
    }
}

// ---------------------------------------------------------------------------
// Sliding-window flash attention. One wave = 16 query rows of one head.
// Q,K: bf16 [S,1024] (head h at col h*64).  Vt: bf16 [1024,S].
// Out: bf16 [S,1024] attention output (pre out_proj).
// Row-sum via extra WMMA against all-ones B; row-max via v_permlane16.
// ---------------------------------------------------------------------------
__global__ __launch_bounds__(256) void attn_kernel(const bf16* __restrict__ Q,
                                                   const bf16* __restrict__ K,
                                                   const bf16* __restrict__ Vt,
                                                   bf16* __restrict__ O) {
  __shared__ bf16 plds[8][16][40];  // wave-private 16x32 P tile, padded stride
  const int lane = threadIdx.x & 31;
  const int wv   = threadIdx.x >> 5;
  const int h    = blockIdx.x;
  const int qt   = blockIdx.y * 128 + wv * 16;
  const int half = lane >> 4, col = lane & 15;

  bf16x16 qa0 = load_a_frag(Q + qt * DMODEL + h * HDIM, DMODEL);
  bf16x16 qa1 = load_a_frag(Q + qt * DMODEL + h * HDIM + 32, DMODEL);

  bf16x16 ones;
  #pragma unroll
  for (int e = 0; e < 16; ++e) ones[e] = (bf16)1.0f;
  f32x8 zc;
  #pragma unroll
  for (int e = 0; e < 8; ++e) zc[e] = 0.f;

  f32x8 oa[4];
  float mr[8], lr[8];
  #pragma unroll
  for (int d = 0; d < 4; ++d)
    #pragma unroll
    for (int e = 0; e < 8; ++e) oa[d][e] = 0.f;
  #pragma unroll
  for (int r = 0; r < 8; ++r) { mr[r] = -1e30f; lr[r] = 0.f; }

  const int kb0 = (qt >= WIN) ? ((qt - (WIN - 1)) & ~31) : 0;
  for (int kb = kb0; kb < qt + 16; kb += 32) {
    // scores S[16 x 32] = Q_tile (16x64) * K_block^T ; chained K-depth 2x32
    bf16x16 k00 = load_b_frag(K + kb * DMODEL + h * HDIM, DMODEL);
    bf16x16 k01 = load_b_frag(K + kb * DMODEL + h * HDIM + 32, DMODEL);
    bf16x16 k10 = load_b_frag(K + (kb + 16) * DMODEL + h * HDIM, DMODEL);
    bf16x16 k11 = load_b_frag(K + (kb + 16) * DMODEL + h * HDIM + 32, DMODEL);
    f32x8 s0 = zc, s1 = zc;
    s0 = WMMA_BF16(qa0, k00, s0);
    s0 = WMMA_BF16(qa1, k01, s0);
    s1 = WMMA_BF16(qa0, k10, s1);
    s1 = WMMA_BF16(qa1, k11, s1);

    // mask only blocks touching the diagonal or the window's lower edge
    // valid iff j <= i && j + WIN > i
    const bool need_mask = (kb + 31 > qt) || (kb < qt - (WIN - 16));
    if (need_mask) {
      #pragma unroll
      for (int r = 0; r < 8; ++r) {
        int i  = qt + r + half * 8;
        int j0 = kb + col;
        int j1 = kb + 16 + col;
        if (!((j0 <= i) && (j0 + WIN > i))) s0[r] = MASK_NEG;
        if (!((j1 <= i) && (j1 + WIN > i))) s1[r] = MASK_NEG;
      }
    }

    // online softmax: row-max via permlane16 butterfly (row = 16-lane half)
    float scv[8];
    #pragma unroll
    for (int r = 0; r < 8; ++r) {
      float t  = rowmax16(fmaxf(s0[r], s1[r]));
      float mn = fmaxf(mr[r], t);
      scv[r] = __expf(mr[r] - mn);
      mr[r]  = mn;
      float p0 = __expf(s0[r] - mn);
      float p1 = __expf(s1[r] - mn);
      // stage P (C-layout -> row-major) through wave-private LDS
      plds[wv][r + half * 8][col]      = (bf16)p0;
      plds[wv][r + half * 8][col + 16] = (bf16)p1;
    }
    asm volatile("s_wait_dscnt 0" ::: "memory");  // wave-local DS ordering

    bf16x16 pa = load_a_frag(&plds[wv][0][0], 40);
    // row-sum of P via the matrix pipe: lsum[r] = sum_k P[r,k]
    f32x8 lsum = WMMA_BF16(pa, ones, zc);
    #pragma unroll
    for (int r = 0; r < 8; ++r) {
      lr[r] = lr[r] * scv[r] + lsum[r];
      oa[0][r] *= scv[r]; oa[1][r] *= scv[r];
      oa[2][r] *= scv[r]; oa[3][r] *= scv[r];
    }
    #pragma unroll
    for (int d = 0; d < 4; ++d) {
      bf16x16 bv = load_b_frag(Vt + (h * HDIM + d * 16) * S_LEN + kb, S_LEN);
      oa[d] = WMMA_BF16(pa, bv, oa[d]);
    }
  }

  #pragma unroll
  for (int r = 0; r < 8; ++r) {
    float inv = 1.f / lr[r];
    int base = (qt + r + half * 8) * DMODEL + h * HDIM + col;
    O[base]      = (bf16)(oa[0][r] * inv);
    O[base + 16] = (bf16)(oa[1][r] * inv);
    O[base + 32] = (bf16)(oa[2][r] * inv);
    O[base + 48] = (bf16)(oa[3][r] * inv);
  }
}

// ---------------------------------------------------------------------------
extern "C" void kernel_launch(void* const* d_in, const int* in_sizes, int n_in,
                              void* d_out, int out_size, void* d_ws, size_t ws_size,
                              hipStream_t stream) {
  const float* X  = (const float*)d_in[0];
  const float* Wq = (const float*)d_in[1];
  const float* Wk = (const float*)d_in[2];
  const float* Wv = (const float*)d_in[3];
  const float* Wo = (const float*)d_in[4];
  const float* bo = (const float*)d_in[5];

  char* ws = (char*)d_ws;
  const size_t MB = 1ull << 20;
  bf16* Xbf = (bf16*)(ws + 0 * MB);   // 8 MB  : X bf16 [4096,1024]
  bf16* Wqt = (bf16*)(ws + 8 * MB);   // 2 MB  : Wq^T bf16 [1024,1024]
  bf16* Wkt = (bf16*)(ws + 10 * MB);
  bf16* Wvt = (bf16*)(ws + 12 * MB);
  bf16* Wot = (bf16*)(ws + 14 * MB);
  bf16* Qb  = (bf16*)(ws + 16 * MB);  // 8 MB  : Q bf16 [4096,1024]
  bf16* Kb  = (bf16*)(ws + 24 * MB);  // 8 MB
  bf16* Vtb = (bf16*)(ws + 32 * MB);  // 8 MB  : V^T bf16 [1024,4096]
  bf16* Ab  = (bf16*)(ws + 40 * MB);  // 8 MB  : attn out bf16 [4096,1024]

  // 1) convert activations to bf16
  cvt_f32_bf16<<<(S_LEN * DMODEL / 4 + 255) / 256, 256, 0, stream>>>(X, Xbf, S_LEN * DMODEL / 4);
  // 2) transpose+convert weights
  dim3 tg(32, 32), tb(32, 8);
  transpose_cvt<<<tg, tb, 0, stream>>>(Wq, Wqt);
  transpose_cvt<<<tg, tb, 0, stream>>>(Wk, Wkt);
  transpose_cvt<<<tg, tb, 0, stream>>>(Wv, Wvt);
  transpose_cvt<<<tg, tb, 0, stream>>>(Wo, Wot);
  // 3) projections (V stored transposed for the PV GEMM's B operand)
  dim3 gg(DMODEL / 128, S_LEN / 128);
  gemm_bf16<0><<<gg, 256, 0, stream>>>(Xbf, Wqt, (void*)Qb, nullptr);
  gemm_bf16<0><<<gg, 256, 0, stream>>>(Xbf, Wkt, (void*)Kb, nullptr);
  gemm_bf16<1><<<gg, 256, 0, stream>>>(Xbf, Wvt, (void*)Vtb, nullptr);
  // 4) sliding-window flash attention
  attn_kernel<<<dim3(NHEADS, S_LEN / 128), 256, 0, stream>>>(Qb, Kb, Vtb, Ab);
  // 5) output projection + bias (f32 out)
  gemm_bf16<2><<<gg, 256, 0, stream>>>(Ab, Wot, d_out, bo);
}
